// NerfModel_53558242181525
// MI455X (gfx1250) — compile-verified, matrix operand order checked by reading, not run on hardware
//
#include <hip/hip_runtime.h>
#include <hip/hip_bf16.h>

// ---------------------------------------------------------------------------
// NeRF forward for MI455X (gfx1250).
// Compute core: v_wmma_f32_16x16x32_f16 GEMMs (f16 in / f32 accum), LDS
// double-buffered with async global->LDS copies (global_load_async_to_lds_b128
// + s_wait_asynccnt). Wave tile 32x64 (2x4 WMMA accumulators) to amortize
// LDS fragment reads and keep the XDL pipe fed.
// ---------------------------------------------------------------------------

#define B_RAYS   1024
#define NSAMP    64
#define NFINE    192          // 64 coarse + 128 pdf samples, sorted
#define MC       (B_RAYS*NSAMP)    // 65536
#define MF       (B_RAYS*NFINE)    // 196608
#define LDH      320          // activation row stride (256 + 63 skip + pad)

typedef __attribute__((ext_vector_type(16))) _Float16 v16h;
typedef __attribute__((ext_vector_type(8)))  _Float16 v8h;
typedef __attribute__((ext_vector_type(8)))  float    v8f;
typedef int v4i_vs __attribute__((vector_size(16)));   // async builtin pointee

#if defined(__has_builtin)
#  if __has_builtin(__builtin_amdgcn_global_load_async_to_lds_b128) && \
      __has_builtin(__builtin_amdgcn_s_wait_asynccnt)
#    define USE_ASYNC 1
#  endif
#endif
#ifndef USE_ASYNC
#  define USE_ASYNC 0
#endif

#if USE_ASYNC
// (v4i as1*, v4i as3*, imm offset, imm cpol)
#define ASYNC_CP16(gsrc, ldst)                                                  \
  __builtin_amdgcn_global_load_async_to_lds_b128(                               \
      (__attribute__((address_space(1))) v4i_vs*)(gsrc),                        \
      (__attribute__((address_space(3))) v4i_vs*)(ldst), 0, 0)
#endif

// ---------------------------------------------------------------------------
// Tiled WMMA GEMM:  C[M x N] = act(A[M x K] @ W[K x N] + bias)
//   A  : f16 row-major, row stride lda
//   Wt : f16, weights pre-transposed+padded: Wt[n*Kpad + k]
//   C  : f16 row-major, row stride ldc
// BM=64, BN=BNT (128 or 256), BK=32. 256 threads = 8 waves, wave grid
// 2(M) x 4(N); each wave computes 32 x (BNT/4) via 2 x (BNT/64) WMMAs/K-step.
// ---------------------------------------------------------------------------
#define BM 64
#define BK 32

template <int RELU, int BNT>
__global__ __launch_bounds__(256) void gemm_f16(
    const _Float16* __restrict__ A, int lda,
    const _Float16* __restrict__ Wt, int ldw,      // ldw == Kpad
    const float* __restrict__ bias,
    _Float16* __restrict__ C, int ldc,
    int M, int K)                                  // K == Kpad (mult of 32)
{
  constexpr int NI  = BNT / 64;    // 16-wide column subtiles per wave (2 or 4)
  constexpr int BCH = BNT / 64;    // v8h chunks of the B tile per thread

  __shared__ __align__(16) _Float16 sA[2][BM * BK];
  __shared__ __align__(16) _Float16 sB[2][BNT * BK];

  const int tid = threadIdx.x;
  const int bm  = blockIdx.x * BM;
  const int bn  = blockIdx.y * BNT;

  // staging maps (all b128 transfers)
  const int a_row = tid >> 2;          // 0..63
  const int a_col = (tid & 3) * 8;     // 0/8/16/24 (f16 elems)
  const _Float16* gA = A + (size_t)(bm + a_row) * lda + a_col;

  __builtin_prefetch(Wt, 0, 1);        // global_prefetch (weights -> L2)

  const int wave = tid >> 5;
  const int lane = tid & 31;
  const int wm   = wave & 1;           // M sub-block of 32
  const int wn   = wave >> 1;          // N sub-block of NI*16
  const int l16  = lane & 15;
  const int kh   = (lane >> 4) * 8;    // hi half-wave takes K+8 / K+24

  v8f acc[2][NI] = {};

  auto stage = [&](int buf, int k0) {
    // A tile: 64x32 f16, one b128 per thread
#if USE_ASYNC
    ASYNC_CP16(gA + k0, &sA[buf][a_row * BK + a_col]);
#else
    *(v8h*)&sA[buf][a_row * BK + a_col] = *(const v8h*)(gA + k0);
#endif
    // B tile: BNT x 32 f16, BCH b128 chunks per thread
#pragma unroll
    for (int j = 0; j < BCH; ++j) {
      const int id = tid * BCH + j;        // chunk id over the tile
      const int br = id >> 2;              // B row (output column)
      const int bc = (id & 3) * 8;         // K offset within row
      const _Float16* src = Wt + (size_t)(bn + br) * ldw + k0 + bc;
#if USE_ASYNC
      ASYNC_CP16(src, &sB[buf][br * BK + bc]);
#else
      *(v8h*)&sB[buf][br * BK + bc] = *(const v8h*)src;
#endif
    }
  };

  const int nk = K / BK;
  stage(0, 0);

  for (int kb = 0; kb < nk; ++kb) {
    const int buf = kb & 1;
#if USE_ASYNC
    __builtin_amdgcn_s_wait_asynccnt(0);
#endif
    __syncthreads();
    if (kb + 1 < nk) stage(buf ^ 1, (kb + 1) * BK);

    const _Float16* psA = sA[buf];
    const _Float16* psB = sB[buf];

    v16h afrag[2], bfrag[NI];
#pragma unroll
    for (int mi = 0; mi < 2; ++mi) {
      const int r = wm * 32 + mi * 16 + l16;
      union { v16h v; v8h h[2]; } u;
      u.h[0] = *(const v8h*)(psA + r * BK + kh);        // K = kh..kh+7
      u.h[1] = *(const v8h*)(psA + r * BK + 16 + kh);   // K = 16+kh..
      afrag[mi] = u.v;
    }
#pragma unroll
    for (int ni = 0; ni < NI; ++ni) {
      const int c = wn * (NI * 16) + ni * 16 + l16;
      union { v16h v; v8h h[2]; } u;
      u.h[0] = *(const v8h*)(psB + c * BK + kh);
      u.h[1] = *(const v8h*)(psB + c * BK + 16 + kh);
      bfrag[ni] = u.v;
    }
#pragma unroll
    for (int mi = 0; mi < 2; ++mi)
#pragma unroll
      for (int ni = 0; ni < NI; ++ni)
        acc[mi][ni] = __builtin_amdgcn_wmma_f32_16x16x32_f16(
            false, afrag[mi], false, bfrag[ni],
            (short)0, acc[mi][ni], false, false);

    __syncthreads();
  }

  // epilogue: D layout = VGPR e -> row (e + 8*hi-half), col l16
#pragma unroll
  for (int ni = 0; ni < NI; ++ni) {
    const int col = bn + wn * (NI * 16) + ni * 16 + l16;
    const float bv = bias[col];
#pragma unroll
    for (int mi = 0; mi < 2; ++mi) {
#pragma unroll
      for (int e = 0; e < 8; ++e) {
        const int row = bm + wm * 32 + mi * 16 + (kh >> 3) * 8 + e;
        if (row < M) {
          float v = acc[mi][ni][e] + bv;
          if (RELU) v = fmaxf(v, 0.f);
          C[(size_t)row * ldc + col] = (_Float16)v;
        }
      }
    }
  }
}

// ---------------------------------------------------------------------------
// Weight prep: W f32 [K][N]  ->  Wt f16 [N][Kpad] (transposed, zero-padded)
// ---------------------------------------------------------------------------
__global__ void prep_weight(const float* __restrict__ W,
                            _Float16* __restrict__ Wt,
                            int K, int N, int Kpad) {
  int i = blockIdx.x * 256 + threadIdx.x;
  if (i >= N * Kpad) return;
  int n = i / Kpad, kp = i - n * Kpad;
  Wt[i] = (kp < K) ? (_Float16)W[(size_t)kp * N + n] : (_Float16)0.f;
}

// ---------------------------------------------------------------------------
// z_vals for the coarse pass: linspace NEAR..FAR over 64 samples
// ---------------------------------------------------------------------------
__global__ void zvals_coarse(float* __restrict__ zv) {
  int i = blockIdx.x * 256 + threadIdx.x;
  if (i >= B_RAYS * NSAMP) return;
  float t = (float)(i % NSAMP) / (float)(NSAMP - 1);
  zv[i] = 2.0f * (1.f - t) + 6.0f * t;
}

// ---------------------------------------------------------------------------
// posenc(point, deg=10) -> 63 features + zero pad, f16 row of 64
// ---------------------------------------------------------------------------
__global__ void posenc_points(const float* __restrict__ rays,
                              const float* __restrict__ zv,
                              _Float16* __restrict__ X0,
                              int S, int M) {
  int m = blockIdx.x * 256 + threadIdx.x;
  if (m >= M) return;
  int ray = m / S, s = m - ray * S;
  float p[3];
  float z = zv[(size_t)ray * S + s];
#pragma unroll
  for (int i = 0; i < 3; ++i)
    p[i] = rays[ray * 6 + i] + z * rays[ray * 6 + 3 + i];
  _Float16* o = X0 + (size_t)m * 64;
#pragma unroll
  for (int i = 0; i < 3; ++i) o[i] = (_Float16)p[i];
#pragma unroll
  for (int l = 0; l < 10; ++l) {
    float sc = (float)(1 << l);
#pragma unroll
    for (int i = 0; i < 3; ++i) {
      float v = p[i] * sc;
      o[3 + l * 3 + i]  = (_Float16)__sinf(v);
      o[33 + l * 3 + i] = (_Float16)__cosf(v);
    }
  }
  o[63] = (_Float16)0.f;
}

// viewdirs_enc: [B_RAYS][27] f32  (dir/||dir||, deg=4)
__global__ void viewdirs_enc_k(const float* __restrict__ rays,
                               float* __restrict__ vde) {
  int r = blockIdx.x * 256 + threadIdx.x;
  if (r >= B_RAYS) return;
  float d[3];
  float n2 = 0.f;
#pragma unroll
  for (int i = 0; i < 3; ++i) { d[i] = rays[r * 6 + 3 + i]; n2 += d[i] * d[i]; }
  float inv = rsqrtf(n2);
  float v[3];
#pragma unroll
  for (int i = 0; i < 3; ++i) { v[i] = d[i] * inv; vde[r * 27 + i] = v[i]; }
#pragma unroll
  for (int l = 0; l < 4; ++l) {
    float sc = (float)(1 << l);
#pragma unroll
    for (int i = 0; i < 3; ++i) {
      float x = v[i] * sc;
      vde[r * 27 + 3 + l * 3 + i]  = __sinf(x);
      vde[r * 27 + 15 + l * 3 + i] = __cosf(x);
    }
  }
}

// skip concat: H[:,256..319] = X0[:,0..63]  (col 63 is already zero)
__global__ void skip_fill(const _Float16* __restrict__ X0,
                          _Float16* __restrict__ H, int M) {
  int i = blockIdx.x * 256 + threadIdx.x;
  if (i >= M * 64) return;
  int m = i >> 6, c = i & 63;
  H[(size_t)m * LDH + 256 + c] = X0[(size_t)m * 64 + c];
}

// condition concat: H[:,256..282] = viewdirs_enc(ray), H[:,283..319] = 0
__global__ void cond_fill(const float* __restrict__ vde,
                          _Float16* __restrict__ H, int S, int M) {
  int i = blockIdx.x * 256 + threadIdx.x;
  if (i >= M * 64) return;
  int m = i >> 6, c = i & 63;
  int ray = m / S;
  H[(size_t)m * LDH + 256 + c] =
      (c < 27) ? (_Float16)vde[ray * 27 + c] : (_Float16)0.f;
}

// alpha head: raw[m][3] = h . wa + ba   (K=256)
__global__ void head_alpha(const _Float16* __restrict__ H, int ldh,
                           const float* __restrict__ wa,
                           const float* __restrict__ ba,
                           float* __restrict__ raw, int M) {
  int m = blockIdx.x * 256 + threadIdx.x;
  if (m >= M) return;
  const _Float16* h = H + (size_t)m * ldh;
  float a = 0.f;
  for (int k = 0; k < 256; ++k) a += (float)h[k] * wa[k];
  raw[(size_t)m * 4 + 3] = a + ba[0];
}

// rgb head: raw[m][0..2] = h . wr + br   (K=128, wr is [128][3])
__global__ void head_rgb(const _Float16* __restrict__ H, int ldh,
                         const float* __restrict__ wr,
                         const float* __restrict__ br,
                         float* __restrict__ raw, int M) {
  int m = blockIdx.x * 256 + threadIdx.x;
  if (m >= M) return;
  const _Float16* h = H + (size_t)m * ldh;
  float a0 = 0.f, a1 = 0.f, a2 = 0.f;
  for (int k = 0; k < 128; ++k) {
    float hv = (float)h[k];
    a0 += hv * wr[k * 3 + 0];
    a1 += hv * wr[k * 3 + 1];
    a2 += hv * wr[k * 3 + 2];
  }
  raw[(size_t)m * 4 + 0] = a0 + br[0];
  raw[(size_t)m * 4 + 1] = a1 + br[1];
  raw[(size_t)m * 4 + 2] = a2 + br[2];
}

// volumetric rendering (one thread per ray, sequential scan over samples)
__global__ void render_k(const float* __restrict__ raw,
                         const float* __restrict__ zv,
                         const float* __restrict__ rays, int S,
                         float* __restrict__ rgb_out,
                         float* __restrict__ disp_out,
                         float* __restrict__ acc_out,
                         float* __restrict__ weights_out) {
  int r = blockIdx.x * 256 + threadIdx.x;
  if (r >= B_RAYS) return;
  float dx = rays[r * 6 + 3], dy = rays[r * 6 + 4], dz = rays[r * 6 + 5];
  float dn = sqrtf(dx * dx + dy * dy + dz * dz);
  float T = 1.f, cr = 0.f, cg = 0.f, cb = 0.f, depth = 0.f, acc = 0.f;
  for (int s = 0; s < S; ++s) {
    const float* rw = raw + ((size_t)r * S + s) * 4;
    float sr = 1.f / (1.f + __expf(-rw[0]));
    float sg = 1.f / (1.f + __expf(-rw[1]));
    float sb = 1.f / (1.f + __expf(-rw[2]));
    float sigma = fmaxf(rw[3], 0.f);
    float z = zv[(size_t)r * S + s];
    float dist = ((s + 1 < S) ? zv[(size_t)r * S + s + 1] - z : 1e10f) * dn;
    float alpha = 1.f - __expf(-sigma * dist);
    float w = alpha * T;
    T *= (1.f - alpha + 1e-10f);
    cr += w * sr; cg += w * sg; cb += w * sb;
    depth += w * z; acc += w;
    if (weights_out) weights_out[(size_t)r * S + s] = w;
  }
  float disp = acc / depth;
  bool ok = (disp > 0.f) && (disp < 1e10f) && (acc > 1e-10f);
  disp = ok ? disp : 1e10f;
  rgb_out[r * 3 + 0] = cr + (1.f - acc);
  rgb_out[r * 3 + 1] = cg + (1.f - acc);
  rgb_out[r * 3 + 2] = cb + (1.f - acc);
  disp_out[r] = disp;
  acc_out[r] = acc;
}

// inverse-CDF sampling of 128 fine z's + merge with coarse 64 -> sorted 192
__global__ void pdf_sample_k(const float* __restrict__ zv_c,
                             const float* __restrict__ wts,
                             float* __restrict__ zv_f) {
  int r = blockIdx.x * 64 + threadIdx.x;
  if (r >= B_RAYS) return;
  const float* zv = zv_c + (size_t)r * NSAMP;
  float bins[63], w[62], cdf[63], zs[128];
  for (int i = 0; i < 63; ++i) bins[i] = 0.5f * (zv[i] + zv[i + 1]);
  float wsum = 0.f;
  for (int i = 0; i < 62; ++i) { w[i] = wts[(size_t)r * NSAMP + 1 + i]; wsum += w[i]; }
  float pad = fmaxf(0.f, 1e-5f - wsum);
  for (int i = 0; i < 62; ++i) w[i] += pad / 62.f;
  wsum += pad;
  cdf[0] = 0.f;
  float c = 0.f;
  for (int i = 0; i < 61; ++i) { c += w[i] / wsum; cdf[i + 1] = fminf(1.f, c); }
  cdf[62] = 1.f;
  for (int j = 0; j < 128; ++j) {
    float u = (float)j / 127.f;
    int i0 = 0, i1 = 62;
    for (int i = 0; i < 63; ++i) if (u >= cdf[i]) i0 = i;
    for (int i = 62; i >= 0; --i) if (cdf[i] > u) i1 = i;
    float denom = cdf[i1] - cdf[i0];
    if (denom < 1e-5f) denom = 1.f;
    float t = (u - cdf[i0]) / denom;
    zs[j] = bins[i0] + t * (bins[i1] - bins[i0]);
  }
  // merge (both sorted)
  float* out = zv_f + (size_t)r * NFINE;
  int a = 0, b = 0;
  for (int k = 0; k < NFINE; ++k) {
    float va = (a < NSAMP) ? zv[a] : 3.4e38f;
    float vb = (b < 128) ? zs[b] : 3.4e38f;
    if (va <= vb) { out[k] = va; ++a; } else { out[k] = vb; ++b; }
  }
}

// ---------------------------------------------------------------------------
// Host-side orchestration
// ---------------------------------------------------------------------------
static inline unsigned cdiv(unsigned a, unsigned b) { return (a + b - 1) / b; }

extern "C" void kernel_launch(void* const* d_in, const int* in_sizes, int n_in,
                              void* d_out, int out_size, void* d_ws, size_t ws_size,
                              hipStream_t stream) {
  (void)in_sizes; (void)n_in; (void)out_size; (void)ws_size;
  // d_in layout (setup_inputs dict order, params dicts flattened in insertion
  // order): [0]=rays, [1..24]=coarse {w0,b0,..,w7,b7,wa,ba,wb,bb,wc0,bc0,wr,br},
  // [25..48]=fine same order.
  const float* rays = (const float*)d_in[0];
  auto P = [&](int set, int idx) -> const float* {
    return (const float*)d_in[1 + set * 24 + idx];
  };

  // ---- workspace carve-up -------------------------------------------------
  char* ws = (char*)d_ws;
  size_t off = 0;
  auto carve = [&](size_t bytes) -> void* {
    void* p = ws + off;
    off = (off + bytes + 255) & ~(size_t)255;
    return p;
  };
  // transposed f16 weights: [set][slot]; slots 0..7 layers, 8=wb, 9=wc0
  static const int wtElems[10] = {256 * 64, 65536, 65536, 65536, 65536,
                                  256 * 320, 65536, 65536, 65536, 128 * 320};
  _Float16* WT[2][10];
  for (int s = 0; s < 2; ++s)
    for (int j = 0; j < 10; ++j)
      WT[s][j] = (_Float16*)carve((size_t)wtElems[j] * 2);

  float*     zv_c  = (float*)carve((size_t)MC * 4);
  float*     zv_f  = (float*)carve((size_t)MF * 4);
  float*     vde   = (float*)carve((size_t)B_RAYS * 27 * 4);
  float*     wts_c = (float*)carve((size_t)MC * 4);
  float*     raw   = (float*)carve((size_t)MF * 4 * 4);
  _Float16*  X0    = (_Float16*)carve((size_t)MF * 64 * 2);
  _Float16*  HA    = (_Float16*)carve((size_t)MF * LDH * 2);
  _Float16*  HB    = (_Float16*)carve((size_t)MF * LDH * 2);

  float* out    = (float*)d_out;
  float* rgb_c  = out;
  float* disp_c = out + 3072;
  float* acc_c  = out + 4096;
  float* rgb_f  = out + 5120;
  float* disp_f = out + 8192;
  float* acc_f  = out + 9216;

  // ---- weight prep (both param sets) --------------------------------------
  struct WCfg { int din; int K; int N; int Kpad; };
  static const WCfg wc[10] = {
      {0, 63, 256, 64},   {2, 256, 256, 256},  {4, 256, 256, 256},
      {6, 256, 256, 256}, {8, 256, 256, 256},  {10, 319, 256, 320},
      {12, 256, 256, 256},{14, 256, 256, 256}, {18, 256, 256, 256},
      {20, 283, 128, 320}};
  for (int s = 0; s < 2; ++s)
    for (int j = 0; j < 10; ++j) {
      int total = wc[j].N * wc[j].Kpad;
      prep_weight<<<cdiv(total, 256), 256, 0, stream>>>(
          P(s, wc[j].din), WT[s][j], wc[j].K, wc[j].N, wc[j].Kpad);
    }

  zvals_coarse<<<cdiv(MC, 256), 256, 0, stream>>>(zv_c);
  viewdirs_enc_k<<<cdiv(B_RAYS, 256), 256, 0, stream>>>(rays, vde);

  auto gemm = [&](const _Float16* A, int lda, _Float16* Wt, int Kpad,
                  const float* bias, _Float16* C, int ldc, int M, int N,
                  bool relu) {
    if (N % 256 == 0) {
      dim3 g(cdiv(M, BM), N / 256);
      if (relu)
        gemm_f16<1, 256><<<g, 256, 0, stream>>>(A, lda, Wt, Kpad, bias, C, ldc, M, Kpad);
      else
        gemm_f16<0, 256><<<g, 256, 0, stream>>>(A, lda, Wt, Kpad, bias, C, ldc, M, Kpad);
    } else {
      dim3 g(cdiv(M, BM), N / 128);
      if (relu)
        gemm_f16<1, 128><<<g, 256, 0, stream>>>(A, lda, Wt, Kpad, bias, C, ldc, M, Kpad);
      else
        gemm_f16<0, 128><<<g, 256, 0, stream>>>(A, lda, Wt, Kpad, bias, C, ldc, M, Kpad);
    }
  };

  auto run_mlp = [&](int set, int M, int S, const float* zv) {
    posenc_points<<<cdiv(M, 256), 256, 0, stream>>>(rays, zv, X0, S, M);
    gemm(X0, 64,  WT[set][0], 64,  P(set, 1),  HA, LDH, M, 256, true);  // L0
    gemm(HA, LDH, WT[set][1], 256, P(set, 3),  HB, LDH, M, 256, true);  // L1
    gemm(HB, LDH, WT[set][2], 256, P(set, 5),  HA, LDH, M, 256, true);  // L2
    gemm(HA, LDH, WT[set][3], 256, P(set, 7),  HB, LDH, M, 256, true);  // L3
    gemm(HB, LDH, WT[set][4], 256, P(set, 9),  HA, LDH, M, 256, true);  // L4
    skip_fill<<<cdiv(M * 64, 256), 256, 0, stream>>>(X0, HA, M);        // concat
    gemm(HA, LDH, WT[set][5], 320, P(set, 11), HB, LDH, M, 256, true);  // L5
    gemm(HB, LDH, WT[set][6], 256, P(set, 13), HA, LDH, M, 256, true);  // L6
    gemm(HA, LDH, WT[set][7], 256, P(set, 15), HB, LDH, M, 256, true);  // L7
    head_alpha<<<cdiv(M, 256), 256, 0, stream>>>(HB, LDH, P(set, 16),
                                                 P(set, 17), raw, M);
    gemm(HB, LDH, WT[set][8], 256, P(set, 19), HA, LDH, M, 256, false); // wb
    cond_fill<<<cdiv(M * 64, 256), 256, 0, stream>>>(vde, HA, S, M);
    gemm(HA, LDH, WT[set][9], 320, P(set, 21), HB, LDH, M, 128, true);  // wc0
    head_rgb<<<cdiv(M, 256), 256, 0, stream>>>(HB, LDH, P(set, 22),
                                               P(set, 23), raw, M);
  };

  // coarse pass
  run_mlp(0, MC, NSAMP, zv_c);
  render_k<<<cdiv(B_RAYS, 256), 256, 0, stream>>>(raw, zv_c, rays, NSAMP,
                                                  rgb_c, disp_c, acc_c, wts_c);
  // fine sampling + fine pass
  pdf_sample_k<<<cdiv(B_RAYS, 64), 64, 0, stream>>>(zv_c, wts_c, zv_f);
  run_mlp(1, MF, NFINE, zv_f);
  render_k<<<cdiv(B_RAYS, 256), 256, 0, stream>>>(raw, zv_f, rays, NFINE,
                                                  rgb_f, disp_f, acc_f, nullptr);
}
